// TransformerBlock_2396591751190
// MI455X (gfx1250) — matrix operand order repaired and checked by reading, not compiled, not run on hardware
//
#include <hip/hip_runtime.h>
#include <hip/hip_bf16.h>

#define IN_C   256
#define OUT_C  128
#define HEADS  4
#define HC     512
#define YSTR   2048   // [q(512) | k(512) | v(512) | skip/agg(512)]

typedef __attribute__((ext_vector_type(16))) __bf16 v16bf;
typedef __attribute__((ext_vector_type(8)))  float  v8f;

union Frag16 { v16bf v; uint4 q[2]; };

// order-preserving float <-> uint encoding for atomicMax-based segment max
__device__ __forceinline__ unsigned enc_f32(float f) {
  unsigned u = __float_as_uint(f);
  return (u & 0x80000000u) ? ~u : (u | 0x80000000u);
}
__device__ __forceinline__ float dec_f32(unsigned k) {
  return (k & 0x80000000u) ? __uint_as_float(k ^ 0x80000000u)
                           : __uint_as_float(~k);
}

// ---------------- WMMA GEMM with async-LDS weight staging ----------------
// C[M x ncols] = A * W^T + bias.  A: [Mpad x K] bf16 row-major (zero-padded
// rows). B: [ncols x K] bf16 row-major => the 64-column slice for this block
// is one contiguous 128*K-byte region, staged into LDS with
// global_load_async_to_lds_b128 (ASYNCcnt) and read back as ds_load_b128.
// Each wave computes a 16(M) x 64(N) strip; 8 waves cover 128 rows per block.
template <int K>
__global__ void __launch_bounds__(256) wmma_gemm_bf16(
    const __bf16* __restrict__ A,
    const __bf16* __restrict__ B,
    const float* __restrict__ bias,
    float* __restrict__ C, int ldc, int mvalid)
{
  extern __shared__ char smem_raw[];
  __bf16* shB = (__bf16*)smem_raw;          // [64][K]

  const int lane = threadIdx.x & 31;
  const int wave = threadIdx.x >> 5;
  const int half = lane >> 4;               // 0: lanes 0-15, 1: lanes 16-31
  const int lr   = lane & 15;
  const int m0 = (blockIdx.x * 8 + wave) * 16;
  const int n0 = blockIdx.y * 64;

  // ---- async-stage the contiguous B slice into LDS ----
  {
    const uint64_t gsrc = (uint64_t)(uintptr_t)(B + (size_t)n0 * K);
    const uint32_t ldst = (uint32_t)(uintptr_t)shB;   // flat low 32b == LDS offset
    const int nchunks = (64 * K * 2) / 16;            // 16B chunks, multiple of 256
    for (int c = threadIdx.x; c < nchunks; c += 256) {
      uint64_t ga = gsrc + (uint32_t)c * 16u;
      uint32_t la = ldst + (uint32_t)c * 16u;
      asm volatile("global_load_async_to_lds_b128 %0, %1, off"
                   :: "v"(la), "v"(ga) : "memory");
    }
    asm volatile("s_wait_asynccnt 0x0" ::: "memory");
    __syncthreads();
  }

  v8f acc[4] = {};
  // A fragment per ISA layout: lanes 0-15 hold K=[0..7],[16..23];
  // lanes 16-31 hold K=[8..15],[24..31].
  const __bf16* arow = A + (size_t)(m0 + lr) * K + half * 8;

  // software pipeline: prefetch next A fragment while issuing WMMAs
  uint4 a0 = *(const uint4*)(arow);
  uint4 a1 = *(const uint4*)(arow + 16);
#pragma unroll
  for (int k0 = 0; k0 < K; k0 += 32) {
    Frag16 a;
    a.q[0] = a0;
    a.q[1] = a1;
    if (k0 + 32 < K) {
      a0 = *(const uint4*)(arow + k0 + 32);
      a1 = *(const uint4*)(arow + k0 + 48);
    }
#pragma unroll
    for (int j = 0; j < 4; ++j) {
      // B fragment: lane holds column n0+j*16+lr, K=[half*8..half*8+7],[16+half*8..16+half*8+7]
      const __bf16* bp = shB + (j * 16 + lr) * K + k0 + half * 8;
      Frag16 b;
      b.q[0] = *(const uint4*)(bp);
      b.q[1] = *(const uint4*)(bp + 16);
      acc[j] = __builtin_amdgcn_wmma_f32_16x16x32_bf16(
          false, a.v, false, b.v, (short)0, acc[j], false, false);
    }
  }

  // C layout: lane lr (+half*16) holds N=n0+j*16+lr, VGPR r -> M = m0+half*8+r
#pragma unroll
  for (int j = 0; j < 4; ++j) {
    const int col = n0 + j * 16 + lr;
    const float bv = bias[col];
#pragma unroll
    for (int r = 0; r < 8; ++r) {
      const int row = m0 + half * 8 + r;
      if (row < mvalid) C[(size_t)row * ldc + col] = acc[j][r] + bv;
    }
  }
}

// ---------------- conversion / init helpers ----------------
__global__ void __launch_bounds__(256) cvt_x_bf16(const float* __restrict__ x,
                                                  __bf16* __restrict__ xb,
                                                  int N, int Npad) {
  int idx = blockIdx.x * 256 + threadIdx.x;
  if (idx >= Npad * IN_C) return;
  int r = idx / IN_C;
  xb[idx] = (r < N) ? (__bf16)x[idx] : (__bf16)0.f;
}

__global__ void __launch_bounds__(256) cvt_wall_bf16(
    const float* __restrict__ Wq, const float* __restrict__ Wk,
    const float* __restrict__ Wv, const float* __restrict__ Ws,
    __bf16* __restrict__ wb) {
  int idx = blockIdx.x * 256 + threadIdx.x;
  if (idx >= 4 * HC * IN_C) return;
  int r = idx / IN_C, c = idx % IN_C;
  const float* src = (r < 512) ? Wq : (r < 1024) ? Wk : (r < 1536) ? Wv : Ws;
  wb[idx] = (__bf16)src[(r & 511) * IN_C + c];
}

__global__ void __launch_bounds__(256) pack_bias(
    const float* __restrict__ bq, const float* __restrict__ bk,
    const float* __restrict__ bv, const float* __restrict__ bs,
    float* __restrict__ bias) {
  int idx = blockIdx.x * 256 + threadIdx.x;
  if (idx >= 4 * HC) return;
  const float* src = (idx < 512) ? bq : (idx < 1024) ? bk : (idx < 1536) ? bv : bs;
  bias[idx] = src[idx & 511];
}

__global__ void __launch_bounds__(256) init_seg(unsigned* __restrict__ mk,
                                                float* __restrict__ den, int n) {
  int idx = blockIdx.x * 256 + threadIdx.x;
  if (idx >= n) return;
  mk[idx] = 0x007FFFFFu;  // enc(-inf)
  den[idx] = 0.f;
}

__global__ void __launch_bounds__(256) zero_f32(float* __restrict__ p, int n) {
  int idx = blockIdx.x * 256 + threadIdx.x;
  if (idx < n) p[idx] = 0.f;
}

__global__ void __launch_bounds__(256) cvt_agg_bf16(const float* __restrict__ y,
                                                    __bf16* __restrict__ ab,
                                                    int N, int Npad) {
  int idx = blockIdx.x * 256 + threadIdx.x;
  if (idx >= Npad * HC) return;
  int r = idx / HC, c = idx % HC;
  ab[idx] = (r < N) ? (__bf16)y[(size_t)r * YSTR + 1536 + c] : (__bf16)0.f;
}

__global__ void __launch_bounds__(256) cvt_wproj_bf16(const float* __restrict__ W,
                                                      __bf16* __restrict__ wb) {
  int idx = blockIdx.x * 256 + threadIdx.x;
  if (idx >= OUT_C * HC) return;
  wb[idx] = (__bf16)W[idx];
}

// ---------------- edge phase (one wave32 per edge) ----------------
__global__ void __launch_bounds__(256) edge_logits_max(
    const int* __restrict__ ei, const float* __restrict__ y,
    float* __restrict__ logits, unsigned* __restrict__ mk, int E)
{
  int wid  = (int)((blockIdx.x * 256u + threadIdx.x) >> 5);
  int lane = threadIdx.x & 31;
  if (wid >= E) return;                    // uniform per wave
  int s = ei[wid];
  int d = ei[E + wid];
  const float* q = y + (size_t)d * YSTR;         // q[dst]
  const float* k = y + (size_t)s * YSTR + 512;   // k[src]
#pragma unroll
  for (int h = 0; h < HEADS; ++h) {
    float p = 0.f;
#pragma unroll
    for (int i = 0; i < 4; ++i) {
      int c = h * OUT_C + lane + 32 * i;   // coalesced across lanes
      p += q[c] * k[c];
    }
    for (int off = 16; off; off >>= 1) p += __shfl_xor(p, off, 32);
    if (lane == 0) {
      float lg = p * 0.088388347648318447f;   // 1/sqrt(128)
      logits[(size_t)wid * HEADS + h] = lg;
      atomicMax(&mk[(size_t)d * HEADS + h], enc_f32(lg));
    }
  }
}

__global__ void __launch_bounds__(256) edge_exp_denom(
    const int* __restrict__ ei, const unsigned* __restrict__ mk,
    float* __restrict__ ev, float* __restrict__ den, int E)
{
  int idx = blockIdx.x * 256 + threadIdx.x;
  if (idx >= E * HEADS) return;
  int e = idx >> 2, h = idx & 3;
  int d = ei[E + e];
  float m = dec_f32(mk[(size_t)d * HEADS + h]);
  float v = __expf(ev[idx] - m);
  ev[idx] = v;                             // logits buffer becomes exp buffer
  unsafeAtomicAdd(&den[(size_t)d * HEADS + h], v);
}

__global__ void __launch_bounds__(256) edge_scatter(
    const int* __restrict__ ei, const float* __restrict__ ev,
    const float* __restrict__ den, float* __restrict__ y, int E)
{
  int wid  = (int)((blockIdx.x * 256u + threadIdx.x) >> 5);
  int lane = threadIdx.x & 31;
  if (wid >= E) return;
  int s = ei[wid], d = ei[E + wid];
  const float* vv  = y + (size_t)s * YSTR + 1024;  // v[src]
  float*       agg = y + (size_t)d * YSTR + 1536;  // skip slice accumulates agg
#pragma unroll
  for (int h = 0; h < HEADS; ++h) {
    float alpha = ev[(size_t)wid * HEADS + h] / den[(size_t)d * HEADS + h];
#pragma unroll
    for (int i = 0; i < 4; ++i) {
      int c = h * OUT_C + lane + 32 * i;
      unsafeAtomicAdd(&agg[c], alpha * vv[c]);     // global_atomic_add_f32
    }
  }
}

// ---------------- batchnorm (training stats) + leaky relu ----------------
// Pass 1: fully-coalesced grid-stride accumulation. Total thread count is a
// multiple of 128, so each lane always sees the same channel (idx & 127).
__global__ void __launch_bounds__(256) bn_partial(const float* __restrict__ o,
                                                  float* __restrict__ sums,  // [2][128]
                                                  int total) {
  int gt = blockIdx.x * 256 + threadIdx.x;
  int stride = gridDim.x * 256;
  float s = 0.f, s2 = 0.f;
  for (int idx = gt; idx < total; idx += stride) {
    float v = o[idx];
    s += v; s2 += v * v;
  }
  __shared__ float sh[512];
  sh[threadIdx.x] = s;
  sh[256 + threadIdx.x] = s2;
  __syncthreads();
  if ((int)threadIdx.x < 128) {
    float ss  = sh[threadIdx.x]       + sh[threadIdx.x + 128];
    float ss2 = sh[256 + threadIdx.x] + sh[384 + threadIdx.x];
    unsafeAtomicAdd(&sums[threadIdx.x], ss);
    unsafeAtomicAdd(&sums[128 + threadIdx.x], ss2);
  }
}

__global__ void bn_finalize(const float* __restrict__ sums,
                            float* __restrict__ mean,
                            float* __restrict__ istd, int N) {
  int c = threadIdx.x;          // 128 threads
  float m = sums[c] / (float)N;
  float var = sums[128 + c] / (float)N - m * m;   // biased variance
  mean[c] = m;
  istd[c] = rsqrtf(var + 1e-5f);
}

__global__ void __launch_bounds__(256) bn_act(
    const float* __restrict__ o, const float* __restrict__ mean,
    const float* __restrict__ istd, const float* __restrict__ gamma,
    const float* __restrict__ beta, float* __restrict__ out, int total) {
  int idx = blockIdx.x * 256 + threadIdx.x;
  if (idx >= total) return;
  int c = idx & (OUT_C - 1);
  float v = (o[idx] - mean[c]) * istd[c] * gamma[c] + beta[c];
  out[idx] = v > 0.f ? v : 0.01f * v;
}

// ---------------- host side ----------------
extern "C" void kernel_launch(void* const* d_in, const int* in_sizes, int n_in,
                              void* d_out, int out_size, void* d_ws, size_t ws_size,
                              hipStream_t stream)
{
  const float* x  = (const float*)d_in[0];
  const int*   ei = (const int*)d_in[1];          // [2, E] (src row, dst row)
  const float* Wq = (const float*)d_in[2];
  const float* bq = (const float*)d_in[3];
  const float* Wk = (const float*)d_in[4];
  const float* bk = (const float*)d_in[5];
  const float* Wv = (const float*)d_in[6];
  const float* bv = (const float*)d_in[7];
  const float* Ws = (const float*)d_in[8];
  const float* bs = (const float*)d_in[9];
  const float* Wp = (const float*)d_in[10];
  const float* bp = (const float*)d_in[11];
  const float* gm = (const float*)d_in[12];
  const float* bt = (const float*)d_in[13];

  const int N    = in_sizes[0] / IN_C;            // 25000
  const int E    = in_sizes[1] / 2;               // 400000
  const int Npad = ((N + 127) / 128) * 128;       // 25088
  const int Mblk = Npad / 128;                    // 196 (8 waves x 16 rows each)

  char* p = (char*)d_ws;
  auto carve = [&](size_t bytes) -> void* {
    void* r = (void*)p;
    p += (bytes + 255) & ~(size_t)255;
    return r;
  };
  __bf16*   xb   = (__bf16*)  carve((size_t)Npad * IN_C * 2);
  __bf16*   wb   = (__bf16*)  carve((size_t)4 * HC * IN_C * 2);
  float*    bias = (float*)   carve((size_t)4 * HC * 4);
  float*    y    = (float*)   carve((size_t)N * YSTR * 4);     // q|k|v|skip(+agg)
  float*    ev   = (float*)   carve((size_t)E * HEADS * 4);    // logits -> exp
  unsigned* mk   = (unsigned*)carve((size_t)N * HEADS * 4);
  float*    den  = (float*)   carve((size_t)N * HEADS * 4);
  __bf16*   ab   = (__bf16*)  carve((size_t)Npad * HC * 2);
  __bf16*   wpb  = (__bf16*)  carve((size_t)OUT_C * HC * 2);
  float*    o2   = (float*)   carve((size_t)N * OUT_C * 4);
  float*    sums = (float*)   carve(2 * OUT_C * 4);
  float*    mean = (float*)   carve(OUT_C * 4);
  float*    istd = (float*)   carve(OUT_C * 4);

  // stage bf16 operands + init segment-softmax / bn state (every call:
  // workspace is not re-initialized between graph replays)
  cvt_x_bf16   <<<(Npad * IN_C + 255) / 256, 256, 0, stream>>>(x, xb, N, Npad);
  cvt_wall_bf16<<<(4 * HC * IN_C + 255) / 256, 256, 0, stream>>>(Wq, Wk, Wv, Ws, wb);
  pack_bias    <<<(4 * HC + 255) / 256, 256, 0, stream>>>(bq, bk, bv, bs, bias);
  init_seg     <<<(N * HEADS + 255) / 256, 256, 0, stream>>>(mk, den, N * HEADS);
  zero_f32     <<<1, 256, 0, stream>>>(sums, 2 * OUT_C);

  // fused QKVS GEMM: y[N x 2048] = x * [Wq;Wk;Wv;Wskip]^T + bias
  wmma_gemm_bf16<IN_C><<<dim3(Mblk, 2048 / 64), 256, 64 * IN_C * 2, stream>>>(
      xb, wb, bias, y, YSTR, N);

  // edge attention (segment softmax via atomics; q/k/v resident in 192MB L2)
  edge_logits_max<<<(E + 7) / 8, 256, 0, stream>>>(ei, y, ev, mk, E);
  edge_exp_denom <<<(E * HEADS + 255) / 256, 256, 0, stream>>>(ei, mk, ev, den, E);
  edge_scatter   <<<(E + 7) / 8, 256, 0, stream>>>(ei, ev, den, y, E);

  // proj GEMM: o2[N x 128] = (agg+skip) * Wproj^T + bproj
  cvt_agg_bf16  <<<(Npad * HC + 255) / 256, 256, 0, stream>>>(y, ab, N, Npad);
  cvt_wproj_bf16<<<(OUT_C * HC + 255) / 256, 256, 0, stream>>>(Wp, wpb);
  wmma_gemm_bf16<HC><<<dim3(Mblk, OUT_C / 64), 256, 64 * HC * 2, stream>>>(
      ab, wpb, bp, o2, OUT_C, N);

  // batchnorm (batch stats) + leaky relu
  bn_partial <<<256, 256, 0, stream>>>(o2, sums, N * OUT_C);
  bn_finalize<<<1, OUT_C, 0, stream>>>(sums, mean, istd, N);
  bn_act     <<<(N * OUT_C + 255) / 256, 256, 0, stream>>>(
      o2, mean, istd, gm, bt, (float*)d_out, N * OUT_C);
}
